// ChebyKANLayer_4011499454818
// MI455X (gfx1250) — matrix operand order repaired and checked by reading, not compiled, not run on hardware
//
#include <hip/hip_runtime.h>

typedef __attribute__((ext_vector_type(16))) _Float16 v16h;
typedef __attribute__((ext_vector_type(8)))  float    v8f;
typedef __attribute__((ext_vector_type(4)))  unsigned int v4u;

#define IN_DIM   512
#define OUT_DIM  512
#define DEG      9          // 9 coefficients (degree 8)
#define DPAD     10         // padded K stride per input (zero at d==9)
#define KTOT     (IN_DIM * DPAD)      // 5120, multiple of 32
#define BM       128
#define BN       128
#define ICHUNK   16
#define KCHUNK   (ICHUNK * DPAD)      // 160
#define NCHUNKS  (KTOT / KCHUNK)      // 32
#define KT_PER_CHUNK (KCHUNK / 32)    // 5
#define NT_TOTAL (OUT_DIM / 16)       // 32

// ---------------------------------------------------------------------------
// Prepass: repack coeffs [i][o][d] (f32) -> f16 B-matrix fragments in ws.
// Fragment (kt, nt) = 32x16 f16 tile in the kernel's own B order (contiguous k):
//   lane = (k-half << 4) | (n & 15); each lane holds 16 consecutive K values
//   (v0..v7, two f16 per dword), stored as 32 contiguous bytes per lane.
// Fragment base = ((kt*32 + nt) * 512) f16; each thread emits one dword.
// ---------------------------------------------------------------------------
__global__ void cheby_pack_b(const float* __restrict__ coeffs,
                             _Float16* __restrict__ bpack) {
    const int tid    = blockIdx.x * 256 + threadIdx.x;  // dword index
    const int frag   = tid >> 8;        // kt*32 + nt
    const int within = tid & 255;       // lane*8 + v
    const int lane   = within >> 3;
    const int v      = within & 7;
    const int kt     = frag >> 5;
    const int nt     = frag & 31;
    const int n      = nt * 16 + (lane & 15);
    const int g      = lane >> 4;

    unsigned outbits = 0;
#pragma unroll
    for (int h = 0; h < 2; ++h) {
        const int klocal = g * 16 + v * 2 + h;
        const int k = kt * 32 + klocal;
        const int i = k / DPAD;
        const int d = k % DPAD;
        float val = 0.0f;
        if (d < DEG) val = coeffs[(size_t)(i * OUT_DIM + n) * DEG + d];
        union { _Float16 hh; unsigned short us; } cv;
        cv.hh = (_Float16)val;
        outbits |= ((unsigned)cv.us) << (16 * h);
    }
    ((unsigned*)bpack)[tid] = outbits;
}

// ---------------------------------------------------------------------------
// Main fused kernel: tanh + Chebyshev recurrence -> double-buffered LDS f16
// basis tile, software-pipelined with the f16 WMMA GEMM (f32 accumulate).
// Block = 256 threads (8 waves), tile M=128 x N=128.
// Wave (mGroup 0..1, nGroup 0..3) owns 4 m-tiles x 2 n-tiles of 16x16.
// ---------------------------------------------------------------------------
__global__ __launch_bounds__(256)
void cheby_wmma_kernel(const float* __restrict__ x,
                       const _Float16* __restrict__ bpack,
                       float* __restrict__ out) {
    __shared__ __align__(16) _Float16 Atile[2][BM * KCHUNK];   // 2 x 40 KB

    const int tid    = threadIdx.x;
    const int lane   = tid & 31;
    const int wid    = tid >> 5;
    const int mGroup = wid & 1;     // 2 M groups of 64 rows (4 m-tiles)
    const int nGroup = wid >> 1;    // 4 N groups of 32 cols (2 n-tiles)
    const int mBase  = blockIdx.x * BM;
    const int nBase  = blockIdx.y * BN;

    v8f acc[4][2];
#pragma unroll
    for (int m = 0; m < 4; ++m)
#pragma unroll
        for (int n = 0; n < 2; ++n)
            acc[m][n] = (v8f){0.f, 0.f, 0.f, 0.f, 0.f, 0.f, 0.f, 0.f};

    // basis-generation mapping: thread -> (row, half of 8 inputs)
    const int brow  = tid & 127;
    const int bhalf = tid >> 7;
    const float* xrow = x + (size_t)(mBase + brow) * IN_DIM + bhalf * 8;

    // compute tanh + Chebyshev recurrence for 8 inputs, store into LDS buffer
    auto gen_basis = [&](v4u r0, v4u r1, _Float16* Ab) {
        union { v4u u; float f[4]; } x0, x1;
        x0.u = r0; x1.u = r1;
        const float xs[8] = {x0.f[0], x0.f[1], x0.f[2], x0.f[3],
                             x1.f[0], x1.f[1], x1.f[2], x1.f[3]};
#pragma unroll
        for (int j = 0; j < 8; ++j) {
            const float t = tanhf(xs[j]);
            union { _Float16 hh[10]; unsigned uu[5]; } pk;
            pk.hh[0] = (_Float16)1.0f;
            pk.hh[1] = (_Float16)t;
            float tp = 1.0f, tc = t;
#pragma unroll
            for (int d = 2; d < DEG; ++d) {
                const float tn = 2.0f * t * tc - tp;
                pk.hh[d] = (_Float16)tn;
                tp = tc; tc = tn;
            }
            pk.hh[9] = (_Float16)0.0f;   // K padding
            // byte offset = brow*320 + icol*20 -> dword aligned
            unsigned* dst = (unsigned*)&Ab[brow * KCHUNK + (bhalf * 8 + j) * DPAD];
#pragma unroll
            for (int q = 0; q < 5; ++q) dst[q] = pk.uu[q];
        }
    };

    // ---- prologue: basis for chunk 0 into buffer 0 ----
    {
        const v4u* xp = (const v4u*)xrow;
        v4u r0 = __builtin_nontemporal_load(xp);
        v4u r1 = __builtin_nontemporal_load(xp + 1);
        gen_basis(r0, r1, &Atile[0][0]);
    }
    __syncthreads();

    for (int chunk = 0; chunk < NCHUNKS; ++chunk) {
        const int buf   = chunk & 1;
        const bool more = (chunk + 1) < NCHUNKS;

        // issue next chunk's x loads early: their HBM latency hides behind
        // the WMMA phase (loads return in order, so B waits are unaffected)
        v4u r0 = {}, r1 = {};
        if (more) {
            const v4u* xp = (const v4u*)(xrow + (chunk + 1) * ICHUNK);
            r0 = __builtin_nontemporal_load(xp);
            r1 = __builtin_nontemporal_load(xp + 1);
        }

        // ---- WMMA phase: 5 k-steps of 32 over this chunk ----
        const _Float16* Ab = &Atile[buf][0];
#pragma unroll
        for (int ktl = 0; ktl < KT_PER_CHUNK; ++ktl) {
            const int kt = chunk * KT_PER_CHUNK + ktl;

            // A fragments (contiguous k order, matching the B packing):
            // lanes 0-15: rows 0-15, K = g*16 + {0..15}
            const int arow = mGroup * 64 + (lane & 15);
            const int akb  = ktl * 32 + ((lane >> 4) << 4);
            v16h afrag[4];
#pragma unroll
            for (int m = 0; m < 4; ++m) {
                union { v16h h; v4u u[2]; } ua;
                const v4u* ap = (const v4u*)&Ab[(arow + m * 16) * KCHUNK + akb];
                ua.u[0] = ap[0];   // K = base .. base+7
                ua.u[1] = ap[1];   // K = base+8 .. base+15  (+16 bytes)
                afrag[m] = ua.h;
            }

            // load BOTH B fragments before the WMMA burst -> staggered waits
            v16h bfrag[2];
#pragma unroll
            for (int n = 0; n < 2; ++n) {
                const int nt = (nBase >> 4) + nGroup * 2 + n;
                const v4u* bp = (const v4u*)(bpack +
                        (((size_t)(kt * NT_TOTAL + nt)) << 9) + lane * 16);
                union { v16h h; v4u u[2]; } ub;
                ub.u[0] = bp[0];
                ub.u[1] = bp[1];
                bfrag[n] = ub.h;
            }

#pragma unroll
            for (int n = 0; n < 2; ++n)
#pragma unroll
                for (int m = 0; m < 4; ++m)
                    acc[m][n] = __builtin_amdgcn_wmma_f32_16x16x32_f16(
                        false, afrag[m], false, bfrag[n], (short)0,
                        acc[m][n], false, false);
        }

        // ---- basis phase for chunk+1 into the other buffer ----
        if (more) gen_basis(r0, r1, &Atile[buf ^ 1][0]);

        __syncthreads();
    }

    // ---- epilogue: C/D layout -> global f32 (non-temporal, keep L2 for B) ----
    const int cl = lane & 15;
    const int cg = lane >> 4;
#pragma unroll
    for (int m = 0; m < 4; ++m)
#pragma unroll
        for (int n = 0; n < 2; ++n)
#pragma unroll
            for (int r = 0; r < 8; ++r) {
                const int row = mBase + mGroup * 64 + m * 16 + cg * 8 + r;
                const int col = nBase + nGroup * 32 + n * 16 + cl;
                __builtin_nontemporal_store(acc[m][n][r],
                                            &out[(size_t)row * OUT_DIM + col]);
            }
}

extern "C" void kernel_launch(void* const* d_in, const int* in_sizes, int n_in,
                              void* d_out, int out_size, void* d_ws, size_t ws_size,
                              hipStream_t stream) {
    const float* x      = (const float*)d_in[0];   // [16384, 512] f32
    const float* coeffs = (const float*)d_in[1];   // [512, 512, 9] f32
    float* out          = (float*)d_out;           // [16384, 512] f32
    _Float16* bpack     = (_Float16*)d_ws;         // 5120*512*2 = 5 MB

    // Prepass: pack coeffs into WMMA-B fragment layout (f16).
    {
        const int total_dwords = (KTOT / 32) * NT_TOTAL * 256;  // 1,310,720
        cheby_pack_b<<<total_dwords / 256, 256, 0, stream>>>(coeffs, bpack);
    }

    // Fused basis-generation + GEMM.
    {
        dim3 grid(16384 / BM, OUT_DIM / BN);   // (128, 4)
        cheby_wmma_kernel<<<grid, 256, 0, stream>>>(x, bpack, out);
    }
}
